// ConvNet_82978768159522
// MI455X (gfx1250) — matrix operand order repaired and checked by reading, not compiled, not run on hardware
//
#include <hip/hip_runtime.h>

// ---------------- types ----------------
typedef __bf16 bf16_t;
typedef bf16_t v16bf __attribute__((ext_vector_type(16)));
typedef bf16_t v8bf  __attribute__((ext_vector_type(8)));
typedef float  v8f   __attribute__((ext_vector_type(8)));

#define N_NODES 10000
#define N_EDGES 320000
#define UDIM    256
#define HDIM    128
#define NLAYERS 3

static __device__ __forceinline__ bf16_t f32_to_bf16(float f) {
    unsigned u = __float_as_uint(f);
    u += 0x7FFFu + ((u >> 16) & 1u);          // round-to-nearest-even
    unsigned short h = (unsigned short)(u >> 16);
    bf16_t r;
    __builtin_memcpy(&r, &h, 2);
    return r;
}

// ---------------- embedding ----------------
__global__ void k_node_embed(const float* __restrict__ pos, const float* __restrict__ Wp,
                             const float* __restrict__ bp, float* __restrict__ x,
                             bf16_t* __restrict__ x_bf) {
    int idx = blockIdx.x * blockDim.x + threadIdx.x;   // N*U threads
    if (idx >= N_NODES * UDIM) return;
    int u = idx & (UDIM - 1);
    int n = idx >> 8;
    float v = pos[n * 2 + 0] * Wp[u] + pos[n * 2 + 1] * Wp[UDIM + u] + bp[u];
    x[idx] = v;
    x_bf[idx] = f32_to_bf16(v);
}

__global__ void k_edge_embed(const float* __restrict__ ea, const float* __restrict__ Wa,
                             const float* __restrict__ ba, float* __restrict__ e,
                             bf16_t* __restrict__ e_bf) {
    int idx = blockIdx.x * blockDim.x + threadIdx.x;   // E*U threads
    if (idx >= N_EDGES * UDIM) return;
    int u = idx & (UDIM - 1);
    int ed = idx >> 8;
    float v = ea[ed] * Wa[u] + ba[u];
    e[idx] = v;
    e_bf[idx] = f32_to_bf16(v);
}

// ---------------- weight pack: fp32 row-major [K x NC] -> bf16 WMMA-B layout ----
// out[((kt*NT + nt)*32 + lane)*16 + v2] = W[kt*32 + (v2>>3)*16 + (lane>>4)*8 + (v2&7)][nt*16 + (lane&15)]
__global__ void k_pack_b(const float* __restrict__ W, bf16_t* __restrict__ out,
                         int K, int NC) {
    int idx = blockIdx.x * blockDim.x + threadIdx.x;   // K*NC threads
    if (idx >= K * NC) return;
    int v2   = idx & 15;
    int lane = (idx >> 4) & 31;
    int t    = idx >> 9;            // kt*NT + nt
    int NT   = NC >> 4;
    int nt   = t % NT;
    int kt   = t / NT;
    int k = kt * 32 + (v2 >> 3) * 16 + (lane >> 4) * 8 + (v2 & 7);
    int c = nt * 16 + (lane & 15);
    out[idx] = f32_to_bf16(W[k * NC + c]);
}

// ---------------- WMMA GEMM: C[M x NC] = A[M x 256](bf16) @ Bpacked ----------
// one wave per block; wave computes a 16-row x 128-col slab (8 accumulators),
// A tile reused across 8 WMMAs per K-step.
__global__ __launch_bounds__(32)
void k_gemm_bf16(const bf16_t* __restrict__ A, const bf16_t* __restrict__ Bp,
                 float* __restrict__ C, int NC) {
    const int lane  = threadIdx.x & 31;
    const int m0    = blockIdx.x * 16;
    const int nt0   = blockIdx.y * 8;      // 8 n-tiles of 16 = 128 cols per wave
    const int NTtot = NC >> 4;
    const int half  = lane >> 4;
    const int rowA  = m0 + (lane & 15);

    v8f acc[8];
    #pragma unroll
    for (int j = 0; j < 8; ++j) acc[j] = (v8f){0.f, 0.f, 0.f, 0.f, 0.f, 0.f, 0.f, 0.f};

    #pragma unroll
    for (int kt = 0; kt < UDIM / 32; ++kt) {
        // A tile (16x32 bf16): two contiguous 16B chunks per lane (ISA A layout)
        const bf16_t* arow = A + (size_t)rowA * UDIM + kt * 32 + half * 8;
        v8bf lo = *(const v8bf*)(arow);
        v8bf hi = *(const v8bf*)(arow + 16);
        v16bf a = __builtin_shufflevector(lo, hi, 0, 1, 2, 3, 4, 5, 6, 7,
                                                  8, 9, 10, 11, 12, 13, 14, 15);
        // B tiles (32x16 bf16 each): 32B contiguous per lane from packed layout
        const bf16_t* bb = Bp + ((size_t)(kt * NTtot + nt0) * 32 + lane) * 16;
        #pragma unroll
        for (int j = 0; j < 8; ++j) {
            v16bf b = *(const v16bf*)(bb + j * 512);
            acc[j] = __builtin_amdgcn_wmma_f32_16x16x32_bf16(
                false, a, false, b, (short)0, acc[j], false, false);
        }
    }
    // C layout: VGPR r -> M = m0 + r + 8*(lane/16), N = lane%16
    const int mrow = m0 + half * 8;
    const int col  = (lane & 15) + nt0 * 16;
    #pragma unroll
    for (int r = 0; r < 8; ++r) {
        float* c = C + (size_t)(mrow + r) * NC + col;
        #pragma unroll
        for (int j = 0; j < 8; ++j) c[j * 16] = acc[j][r];
    }
}

// ---------------- per-layer fused kernels ----------------
__global__ void k_zero(float* __restrict__ p, int n) {
    int idx = blockIdx.x * blockDim.x + threadIdx.x;
    if (idx < n) p[idx] = 0.f;
}

// msg = sigmoid(e_old) * (XW2[src] + b2);  agg[dst] += msg
__global__ void k_msg_scatter(const float* __restrict__ e_old, const float* __restrict__ XW2,
                              const float* __restrict__ b2, const int* __restrict__ src,
                              const int* __restrict__ dst, float* __restrict__ agg) {
    int idx = blockIdx.x * blockDim.x + threadIdx.x;   // E*U threads
    if (idx >= N_EDGES * UDIM) return;
    int u = idx & (UDIM - 1);
    int ed = idx >> 8;
    float g = 1.f / (1.f + __expf(-e_old[idx]));
    float m = g * (XW2[src[ed] * UDIM + u] + b2[u]);
    atomicAdd(&agg[dst[ed] * UDIM + u], m);
}

// x_new = x + relu(XW1raw + b1 + agg); also refresh bf16 copy
__global__ void k_node_epi(const float* __restrict__ x, const float* __restrict__ XW1,
                           const float* __restrict__ b1, const float* __restrict__ agg,
                           float* __restrict__ x_new, bf16_t* __restrict__ x_bf) {
    int idx = blockIdx.x * blockDim.x + threadIdx.x;   // N*U threads
    if (idx >= N_NODES * UDIM) return;
    int u = idx & (UDIM - 1);
    float t = XW1[idx] + b1[u] + agg[idx];
    t = t > 0.f ? t : 0.f;
    float v = x[idx] + t;
    x_new[idx] = v;
    x_bf[idx] = f32_to_bf16(v);
}

// e_io holds raw e@We on entry; e_new = e_old + relu(raw + XWs[src] + XWt[dst] + be)
__global__ void k_edge_epi(const float* __restrict__ e_old, float* __restrict__ e_io,
                           const float* __restrict__ XWs, const float* __restrict__ XWt,
                           const float* __restrict__ be, const int* __restrict__ src,
                           const int* __restrict__ dst, bf16_t* __restrict__ e_bf) {
    int idx = blockIdx.x * blockDim.x + threadIdx.x;   // E*U threads
    if (idx >= N_EDGES * UDIM) return;
    int u = idx & (UDIM - 1);
    int ed = idx >> 8;
    float t = e_io[idx] + XWs[src[ed] * UDIM + u] + XWt[dst[ed] * UDIM + u] + be[u];
    t = t > 0.f ? t : 0.f;
    float v = e_old[idx] + t;
    e_io[idx] = v;
    e_bf[idx] = f32_to_bf16(v);
}

// final head: h = Hraw + a*Wm1[U] + bm1 ; PReLU ; out = h @ Wm2 + bm2 (wave/edge)
__global__ void k_mlp_out(const float* __restrict__ Hraw, const float* __restrict__ ea,
                          const float* __restrict__ wm1_last, const float* __restrict__ bm1,
                          const float* __restrict__ alpha, const float* __restrict__ Wm2,
                          const float* __restrict__ bm2, float* __restrict__ out) {
    int ed = blockIdx.x * (blockDim.x >> 5) + (threadIdx.x >> 5);
    int lane = threadIdx.x & 31;
    if (ed >= N_EDGES) return;
    float a  = ea[ed];
    float al = alpha[0];
    float s = 0.f;
    #pragma unroll
    for (int q = 0; q < HDIM / 32; ++q) {
        int j = q * 32 + lane;
        float h = Hraw[(size_t)ed * HDIM + j] + a * wm1_last[j] + bm1[j];
        h = h > 0.f ? h : al * h;
        s += h * Wm2[j];
    }
    #pragma unroll
    for (int off = 16; off > 0; off >>= 1) s += __shfl_xor(s, off, 32);
    if (lane == 0) out[ed] = s + bm2[0];
}

// ---------------- host orchestration ----------------
extern "C" void kernel_launch(void* const* d_in, const int* in_sizes, int n_in,
                              void* d_out, int out_size, void* d_ws, size_t ws_size,
                              hipStream_t stream) {
    (void)in_sizes; (void)n_in; (void)out_size; (void)ws_size;
    const float* pos  = (const float*)d_in[0];
    const float* ea   = (const float*)d_in[1];
    const int*   eidx = (const int*)  d_in[2];
    const float* Wp   = (const float*)d_in[3];
    const float* bp   = (const float*)d_in[4];
    const float* Wa   = (const float*)d_in[5];
    const float* ba   = (const float*)d_in[6];
    const float* W1   = (const float*)d_in[7];
    const float* b1   = (const float*)d_in[8];
    const float* W2   = (const float*)d_in[9];
    const float* b2   = (const float*)d_in[10];
    const float* We   = (const float*)d_in[11];
    const float* be   = (const float*)d_in[12];
    const float* Ws   = (const float*)d_in[13];
    const float* Wt   = (const float*)d_in[14];
    const float* Wm1  = (const float*)d_in[15];
    const float* bm1  = (const float*)d_in[16];
    const float* alpha= (const float*)d_in[17];
    const float* Wm2  = (const float*)d_in[18];
    const float* bm2  = (const float*)d_in[19];
    float* out = (float*)d_out;

    const int* src = eidx;
    const int* dst = eidx + N_EDGES;

    // workspace carve-up (256B aligned)
    char* ws = (char*)d_ws;
    size_t off = 0;
    auto carve = [&](size_t bytes) -> void* {
        void* p = ws + off;
        off += (bytes + 255) & ~(size_t)255;
        return p;
    };
    float*  xA   = (float*) carve((size_t)N_NODES * UDIM * 4);
    float*  xB   = (float*) carve((size_t)N_NODES * UDIM * 4);
    bf16_t* x_bf = (bf16_t*)carve((size_t)N_NODES * UDIM * 2);
    float*  eA   = (float*) carve((size_t)N_EDGES * UDIM * 4);
    float*  eB   = (float*) carve((size_t)N_EDGES * UDIM * 4);
    bf16_t* e_bf = (bf16_t*)carve((size_t)N_EDGES * UDIM * 2);
    float*  XW1  = (float*) carve((size_t)N_NODES * UDIM * 4);
    float*  XW2  = (float*) carve((size_t)N_NODES * UDIM * 4);
    float*  XWs  = (float*) carve((size_t)N_NODES * UDIM * 4);
    float*  XWt  = (float*) carve((size_t)N_NODES * UDIM * 4);
    float*  agg  = (float*) carve((size_t)N_NODES * UDIM * 4);
    bf16_t* pW1  = (bf16_t*)carve((size_t)UDIM * UDIM * 2);
    bf16_t* pW2  = (bf16_t*)carve((size_t)UDIM * UDIM * 2);
    bf16_t* pWs  = (bf16_t*)carve((size_t)UDIM * UDIM * 2);
    bf16_t* pWt  = (bf16_t*)carve((size_t)UDIM * UDIM * 2);
    bf16_t* pWe  = (bf16_t*)carve((size_t)UDIM * UDIM * 2);
    bf16_t* pWm1 = (bf16_t*)carve((size_t)UDIM * HDIM * 2);

    const int NU = N_NODES * UDIM;                 // 2,560,000
    const int EU = N_EDGES * UDIM;                 // 81,920,000
    const dim3 blk(256);
    const dim3 gNU((NU + 255) / 256), gEU((EU + 255) / 256);

    auto pack = [&](const float* W, bf16_t* dstp, int K, int NC) {
        int tot = K * NC;
        k_pack_b<<<dim3((tot + 255) / 256), blk, 0, stream>>>(W, dstp, K, NC);
    };
    auto gemm = [&](const bf16_t* A, const bf16_t* Bp, float* C, int M, int NC) {
        k_gemm_bf16<<<dim3(M / 16, NC / 128), dim3(32), 0, stream>>>(A, Bp, C, NC);
    };

    // embeddings
    k_node_embed<<<gNU, blk, 0, stream>>>(pos, Wp, bp, xA, x_bf);
    k_edge_embed<<<gEU, blk, 0, stream>>>(ea, Wa, ba, eA, e_bf);

    float *x_cur = xA, *x_nxt = xB, *e_cur = eA, *e_nxt = eB;
    for (int l = 0; l < NLAYERS; ++l) {
        const size_t WO = (size_t)l * UDIM * UDIM;
        pack(W1 + WO, pW1, UDIM, UDIM);
        pack(W2 + WO, pW2, UDIM, UDIM);
        pack(Ws + WO, pWs, UDIM, UDIM);
        pack(Wt + WO, pWt, UDIM, UDIM);
        pack(We + WO, pWe, UDIM, UDIM);

        gemm(x_bf, pW1, XW1, N_NODES, UDIM);
        gemm(x_bf, pW2, XW2, N_NODES, UDIM);
        gemm(x_bf, pWs, XWs, N_NODES, UDIM);
        gemm(x_bf, pWt, XWt, N_NODES, UDIM);
        gemm(e_bf, pWe, e_nxt, N_EDGES, UDIM);     // raw e@We into ping-pong buffer

        k_zero<<<gNU, blk, 0, stream>>>(agg, NU);
        k_msg_scatter<<<gEU, blk, 0, stream>>>(e_cur, XW2, b2 + (size_t)l * UDIM,
                                               src, dst, agg);
        k_node_epi<<<gNU, blk, 0, stream>>>(x_cur, XW1, b1 + (size_t)l * UDIM,
                                            agg, x_nxt, x_bf);
        k_edge_epi<<<gEU, blk, 0, stream>>>(e_cur, e_nxt, XWs, XWt,
                                            be + (size_t)l * UDIM, src, dst, e_bf);
        float* t;
        t = x_cur; x_cur = x_nxt; x_nxt = t;
        t = e_cur; e_cur = e_nxt; e_nxt = t;
    }

    // final edge update (layer L)
    {
        const size_t WO = (size_t)NLAYERS * UDIM * UDIM;
        pack(We + WO, pWe, UDIM, UDIM);
        pack(Ws + WO, pWs, UDIM, UDIM);
        pack(Wt + WO, pWt, UDIM, UDIM);
        gemm(x_bf, pWs, XWs, N_NODES, UDIM);
        gemm(x_bf, pWt, XWt, N_NODES, UDIM);
        gemm(e_bf, pWe, e_nxt, N_EDGES, UDIM);
        k_edge_epi<<<gEU, blk, 0, stream>>>(e_cur, e_nxt, XWs, XWt,
                                            be + (size_t)NLAYERS * UDIM, src, dst, e_bf);
        float* t = e_cur; e_cur = e_nxt; e_nxt = t;
    }

    // head: Hraw = e @ Wm1[0:U]  (reuse the free e ping-pong buffer)
    float* Hraw = e_nxt;
    pack(Wm1, pWm1, UDIM, HDIM);
    gemm(e_bf, pWm1, Hraw, N_EDGES, HDIM);
    k_mlp_out<<<dim3(N_EDGES / 8), blk, 0, stream>>>(Hraw, ea, Wm1 + (size_t)UDIM * HDIM,
                                                     bm1, alpha, Wm2, bm2, out);
}
